// Net_46170898432228
// MI455X (gfx1250) — matrix operand order repaired and checked by reading, not compiled, not run on hardware
//
#include <hip/hip_runtime.h>
#include <math.h>

// ---------------- problem sizes ----------------
#define B_SZ   512
#define S_SZ   128
#define IN_SZ  773
#define K_IH   800      // IN padded to multiple of 32
#define BERT   768
#define VIS    768
#define LAY    64
#define CAT    1600     // BERT + VIS + LAY
#define HID    100

typedef __attribute__((ext_vector_type(16))) __bf16 v16bf;
typedef __attribute__((ext_vector_type(8)))  __bf16 bf16x8;
typedef __attribute__((ext_vector_type(8)))  float  v8f;

// gfx1250 has a hardware v_tanh_f32 (TRANS op); fall back to OCML otherwise.
__device__ inline float fast_tanh(float x) {
#if __has_builtin(__builtin_amdgcn_tanhf)
    return __builtin_amdgcn_tanhf(x);
#elif __has_builtin(__builtin_amdgcn_tanh_f32)
    return __builtin_amdgcn_tanh_f32(x);
#else
    return tanhf(x);
#endif
}

// Scheduling fence: keep the fragment-load group issued before the WMMA group
// so the waits become staged (loadcnt<=N) instead of load->wait0->wmma chains.
__device__ inline void sched_fence() {
#if __has_builtin(__builtin_amdgcn_sched_barrier)
    __builtin_amdgcn_sched_barrier(0);
#endif
}

// ------------- WMMA fragment loaders (bf16, 16x16x32) -------------
// A (16xK tile, row-major). Lane l: row l&15, elems 0..7 = K koff..koff+7,
// elems 8..15 = K koff+16..koff+23, koff=(l>>4)*8.  (ISA 7.12.2)
__device__ inline v16bf load_a_frag(const __bf16* base, int ldk, int lane) {
    int r    = lane & 15;
    int koff = (lane >> 4) << 3;
    const __bf16* p = base + (size_t)r * ldk + koff;
    bf16x8 lo = *(const bf16x8*)p;
    bf16x8 hi = *(const bf16x8*)(p + 16);
    return __builtin_shufflevector(lo, hi, 0,1,2,3,4,5,6,7,8,9,10,11,12,13,14,15);
}

// B operand for D = A * W^T: feed W row-major (N x K). Lane l: col l&15,
// elems 0..7 = K koff..koff+7, elems 8..15 = K koff+16..koff+23, koff=(l>>4)*8.
__device__ inline v16bf load_b_frag(const __bf16* base, int ldk, int lane) {
    int n    = lane & 15;
    int koff = (lane >> 4) << 3;
    const __bf16* p = base + (size_t)n * ldk + koff;
    bf16x8 lo = *(const bf16x8*)p;
    bf16x8 hi = *(const bf16x8*)(p + 16);
    return __builtin_shufflevector(lo, hi, 0,1,2,3,4,5,6,7,8,9,10,11,12,13,14,15);
}

// ---------------- weight conversion kernels ----------------
__global__ void cvt_wih_kernel(const float* __restrict__ w, __bf16* __restrict__ o) {
    int i = blockIdx.x * blockDim.x + threadIdx.x;          // over 768*800
    if (i >= BERT * K_IH) return;
    int n = i / K_IH, k = i % K_IH;
    o[i] = (k < IN_SZ) ? (__bf16)w[(size_t)n * IN_SZ + k] : (__bf16)0.0f;
}

__global__ void cvt_dense_kernel(const float* __restrict__ w, __bf16* __restrict__ o, int n) {
    int i = blockIdx.x * blockDim.x + threadIdx.x;
    if (i < n) o[i] = (__bf16)w[i];
}

// fill concat[:, 768:1600] with bf16(visual) / bf16(layout)
__global__ void cvt_concat_kernel(const float* __restrict__ visual,
                                  const float* __restrict__ layout,
                                  __bf16* __restrict__ concat) {
    int i = blockIdx.x * blockDim.x + threadIdx.x;          // over 512*832
    if (i >= B_SZ * (VIS + LAY)) return;
    int b = i / (VIS + LAY), c = i % (VIS + LAY);
    float v = (c < VIS) ? visual[(size_t)b * VIS + c] : layout[(size_t)b * LAY + (c - VIS)];
    concat[(size_t)b * CAT + BERT + c] = (__bf16)v;
}

// ---------------- GEMM 1: pre[t][b][n] = X @ W_ih^T + (b_ih + b_hh) ----------------
// grid: 65536/16 = 4096 blocks x 256 threads (8 waves). One 16-row M-tile per
// block; N = 768 = 8 waves * 6 tiles. X fp32->bf16 via double-buffered LDS tile
// with software-pipelined staging loads (one barrier per K-chunk).
__global__ void __launch_bounds__(256) gemm_pre_kernel(
        const float* __restrict__ x, const __bf16* __restrict__ wih,
        const float* __restrict__ b_ih, const float* __restrict__ b_hh,
        float* __restrict__ pre) {
    __shared__ __bf16 Atile[2][16][32];
    const int tid  = threadIdx.x;
    const int lane = tid & 31;
    const int wave = tid >> 5;
    const int m0   = blockIdx.x * 16;                        // row in [0,65536)
    const int b    = m0 >> 7;                                // x row = b*128 + t
    const int t0   = m0 & 127;
    const int sr   = (tid * 2) >> 5;                         // staging row
    const int sc   = (tid * 2) & 31;                         // staging col (even)
    const float* xrow = x + (size_t)(m0 + sr) * IN_SZ;

    v8f acc[6];
    #pragma unroll
    for (int i = 0; i < 6; ++i) acc[i] = (v8f)(0.0f);

    // stage chunk 0 (always full)
    float xv0 = xrow[sc], xv1 = xrow[sc + 1];

    const int NKC = K_IH / 32;                               // 25
    for (int kc = 0; kc < NKC; ++kc) {
        const int cur   = kc & 1;
        const int kbase = kc * 32;
        Atile[cur][sr][sc]     = (__bf16)xv0;
        Atile[cur][sr][sc + 1] = (__bf16)xv1;
        __syncthreads();

        // issue next chunk's staging loads early (hidden behind B-loads + WMMAs)
        if (kc + 1 < NKC) {
            const int kb2 = kbase + 32;
            if (kc + 1 < NKC - 1) {
                xv0 = xrow[kb2 + sc];
                xv1 = xrow[kb2 + sc + 1];
            } else {                                         // K tail (768..772 valid)
                xv0 = (kb2 + sc     < IN_SZ) ? xrow[kb2 + sc]     : 0.0f;
                xv1 = (kb2 + sc + 1 < IN_SZ) ? xrow[kb2 + sc + 1] : 0.0f;
            }
        }

        v16bf a = load_a_frag(&Atile[cur][0][0], 32, lane);
        v16bf bfr[6];
        #pragma unroll
        for (int i = 0; i < 6; ++i) {
            int n0 = (wave * 6 + i) * 16;
            bfr[i] = load_b_frag(wih + (size_t)n0 * K_IH + kbase, K_IH, lane);
        }
        sched_fence();
        #pragma unroll
        for (int i = 0; i < 6; ++i)
            acc[i] = __builtin_amdgcn_wmma_f32_16x16x32_bf16(
                         false, a, false, bfr[i], (short)0, acc[i], false, false);
    }

    const int mhalf = (lane >> 4) * 8;
    #pragma unroll
    for (int i = 0; i < 6; ++i) {
        int n = (wave * 6 + i) * 16 + (lane & 15);
        float bias = b_ih[n] + b_hh[n];
        #pragma unroll
        for (int r = 0; r < 8; ++r) {
            int tt = t0 + mhalf + r;                          // time index
            pre[((size_t)tt * B_SZ + b) * BERT + n] = acc[i][r] + bias;
        }
    }
}

// ---------------- RNN recurrence: persistent per 16-row batch tile ----------------
// grid: 32 blocks x 512 threads (16 waves, 3 N-tiles each). h double-buffered as
// bf16 in LDS -> one barrier per step; batch rows independent -> no grid sync.
__global__ void __launch_bounds__(512) rnn_kernel(
        const float* __restrict__ pre, const __bf16* __restrict__ whh,
        __bf16* __restrict__ concat) {
    __shared__ __bf16 hbuf[2][16][BERT];                      // 48 KB
    const int tid   = threadIdx.x;
    const int lane  = tid & 31;
    const int wave  = tid >> 5;                               // 0..15
    const int b0    = blockIdx.x * 16;
    const int mhalf = (lane >> 4) * 8;

    for (int i = tid; i < 16 * BERT; i += 512) (&hbuf[0][0][0])[i] = (__bf16)0.0f;
    __syncthreads();

    for (int t = 0; t < S_SZ; ++t) {
        const int cur = t & 1, nxt = cur ^ 1;
        const float* pb = pre + ((size_t)t * B_SZ + b0) * BERT;

        // prefetch next step's pre tile (48 KB contiguous) while we compute
        if (t + 1 < S_SZ)
            __builtin_prefetch(pre + ((size_t)(t + 1) * B_SZ + b0) * BERT + tid * 24, 0, 1);

        // init accumulators from the precomputed input projection
        v8f acc[3];
        #pragma unroll
        for (int i = 0; i < 3; ++i) {
            int n = (wave * 3 + i) * 16 + (lane & 15);
            #pragma unroll
            for (int r = 0; r < 8; ++r)
                acc[i][r] = pb[(size_t)(mhalf + r) * BERT + n];
        }

        // h @ W_hh^T  (K = 768 = 24 chunks of 32)
        for (int kc = 0; kc < BERT / 32; ++kc) {
            const int kbase = kc * 32;
            v16bf a = load_a_frag(&hbuf[cur][0][kbase], BERT, lane);
            v16bf bfr[3];
            #pragma unroll
            for (int i = 0; i < 3; ++i) {
                int n0 = (wave * 3 + i) * 16;
                bfr[i] = load_b_frag(whh + (size_t)n0 * BERT + kbase, BERT, lane);
            }
            sched_fence();
            #pragma unroll
            for (int i = 0; i < 3; ++i)
                acc[i] = __builtin_amdgcn_wmma_f32_16x16x32_bf16(
                             false, a, false, bfr[i], (short)0, acc[i], false, false);
        }

        // tanh -> write into the *other* buffer (no conflict with readers of cur)
        #pragma unroll
        for (int i = 0; i < 3; ++i) {
            int n = (wave * 3 + i) * 16 + (lane & 15);
            #pragma unroll
            for (int r = 0; r < 8; ++r)
                hbuf[nxt][mhalf + r][n] = (__bf16)fast_tanh(acc[i][r]);
        }
        __syncthreads();                                      // h ready for next step
    }

    // final hidden state lives in hbuf[S_SZ & 1] -> concat[:, 0:768]
    const __bf16* hf = &hbuf[S_SZ & 1][0][0];
    for (int i = tid; i < 16 * BERT; i += 512) {
        int r = i / BERT, c = i % BERT;
        concat[(size_t)(b0 + r) * CAT + c] = hf[i];
    }
}

// ---------------- GEMM 2: combined = relu(concat @ W_lin^T + b_lin) ----------------
// grid: 512/16 = 32 blocks x 256 threads. A already bf16 in global -> direct frags.
__global__ void __launch_bounds__(256) gemm_comb_kernel(
        const __bf16* __restrict__ concat, const __bf16* __restrict__ wlin,
        const float* __restrict__ b_lin, float* __restrict__ combined) {
    const int tid  = threadIdx.x;
    const int lane = tid & 31;
    const int wave = tid >> 5;
    const int m0   = blockIdx.x * 16;

    v8f acc[6];
    #pragma unroll
    for (int i = 0; i < 6; ++i) acc[i] = (v8f)(0.0f);

    for (int kc = 0; kc < CAT / 32; ++kc) {
        const int kbase = kc * 32;
        v16bf a = load_a_frag(concat + (size_t)m0 * CAT + kbase, CAT, lane);
        v16bf bfr[6];
        #pragma unroll
        for (int i = 0; i < 6; ++i) {
            int n0 = (wave * 6 + i) * 16;
            bfr[i] = load_b_frag(wlin + (size_t)n0 * CAT + kbase, CAT, lane);
        }
        sched_fence();
        #pragma unroll
        for (int i = 0; i < 6; ++i)
            acc[i] = __builtin_amdgcn_wmma_f32_16x16x32_bf16(
                         false, a, false, bfr[i], (short)0, acc[i], false, false);
    }

    const int mhalf = (lane >> 4) * 8;
    #pragma unroll
    for (int i = 0; i < 6; ++i) {
        int n = (wave * 6 + i) * 16 + (lane & 15);
        float bias = b_lin[n];
        #pragma unroll
        for (int r = 0; r < 8; ++r) {
            float v = acc[i][r] + bias;
            combined[(size_t)(m0 + mhalf + r) * BERT + n] = v > 0.0f ? v : 0.0f;
        }
    }
}

// ---------------- tiny MLP head 768->100->100->100 (fp32) ----------------
__global__ void __launch_bounds__(128) mlp_kernel(
        const float* __restrict__ combined,
        const float* __restrict__ W1, const float* __restrict__ b1,
        const float* __restrict__ W2, const float* __restrict__ b2,
        const float* __restrict__ Wo, const float* __restrict__ bo,
        float* __restrict__ out) {
    __shared__ float cbuf[BERT];
    __shared__ float h1s[HID];
    __shared__ float h2s[HID];
    const int b = blockIdx.x, tid = threadIdx.x;

    for (int i = tid; i < BERT; i += 128) cbuf[i] = combined[(size_t)b * BERT + i];
    __syncthreads();
    if (tid < HID) {
        float s = b1[tid];
        const float* wr = W1 + (size_t)tid * BERT;
        for (int k = 0; k < BERT; ++k) s += cbuf[k] * wr[k];
        h1s[tid] = s > 0.0f ? s : 0.0f;
    }
    __syncthreads();
    if (tid < HID) {
        float s = b2[tid];
        const float* wr = W2 + (size_t)tid * HID;
        for (int k = 0; k < HID; ++k) s += h1s[k] * wr[k];
        h2s[tid] = s > 0.0f ? s : 0.0f;
    }
    __syncthreads();
    if (tid < HID) {
        float s = bo[tid];
        const float* wr = Wo + (size_t)tid * HID;
        for (int k = 0; k < HID; ++k) s += h2s[k] * wr[k];
        out[(size_t)b * HID + tid] = s;
    }
}

// ---------------- host side ----------------
extern "C" void kernel_launch(void* const* d_in, const int* in_sizes, int n_in,
                              void* d_out, int out_size, void* d_ws, size_t ws_size,
                              hipStream_t stream) {
    const float* x      = (const float*)d_in[0];
    const float* visual = (const float*)d_in[1];
    const float* layout = (const float*)d_in[2];
    const float* W_ih   = (const float*)d_in[3];
    const float* b_ih   = (const float*)d_in[4];
    const float* W_hh   = (const float*)d_in[5];
    const float* b_hh   = (const float*)d_in[6];
    const float* W_lin  = (const float*)d_in[7];
    const float* b_lin  = (const float*)d_in[8];
    const float* W1     = (const float*)d_in[9];
    const float* b1     = (const float*)d_in[10];
    const float* W2     = (const float*)d_in[11];
    const float* b2     = (const float*)d_in[12];
    const float* Wo     = (const float*)d_in[13];
    const float* bo     = (const float*)d_in[14];
    float* out = (float*)d_out;

    char* w = (char*)d_ws;
    __bf16* wih_b  = (__bf16*)w;  w += (size_t)BERT * K_IH * 2;
    __bf16* whh_b  = (__bf16*)w;  w += (size_t)BERT * BERT * 2;
    __bf16* wlin_b = (__bf16*)w;  w += (size_t)BERT * CAT * 2;
    float*  pre    = (float*)w;   w += (size_t)B_SZ * S_SZ * BERT * 4;   // [S][B][768]
    __bf16* concat = (__bf16*)w;  w += (size_t)B_SZ * CAT * 2;
    float*  comb   = (float*)w;   w += (size_t)B_SZ * BERT * 4;

    // weight / feature conversion to bf16
    cvt_wih_kernel   <<<(BERT * K_IH + 255) / 256, 256, 0, stream>>>(W_ih, wih_b);
    cvt_dense_kernel <<<(BERT * BERT + 255) / 256, 256, 0, stream>>>(W_hh, whh_b, BERT * BERT);
    cvt_dense_kernel <<<(BERT * CAT  + 255) / 256, 256, 0, stream>>>(W_lin, wlin_b, BERT * CAT);
    cvt_concat_kernel<<<(B_SZ * (VIS + LAY) + 255) / 256, 256, 0, stream>>>(visual, layout, concat);

    // time-parallel input projection (the big WMMA GEMM)
    gemm_pre_kernel<<<(B_SZ * S_SZ) / 16, 256, 0, stream>>>(x, wih_b, b_ih, b_hh, pre);

    // sequential recurrence: batch-parallel persistent WMMA kernel
    rnn_kernel<<<B_SZ / 16, 512, 0, stream>>>(pre, whh_b, concat);

    // fusion layer
    gemm_comb_kernel<<<B_SZ / 16, 256, 0, stream>>>(concat, wlin_b, b_lin, comb);

    // tiny MLP head
    mlp_kernel<<<B_SZ, 128, 0, stream>>>(comb, W1, b1, W2, b2, Wo, bo, out);

    (void)in_sizes; (void)n_in; (void)out_size; (void)ws_size;
}
